// AdditiveAttention_21131239096573
// MI455X (gfx1250) — compile-verified, matrix operand order checked by reading, not run on hardware
//
#include <hip/hip_runtime.h>
#include <hip/hip_bf16.h>

#define HDIM 512
#define SDIM 2048
#define BDIM 32
#define SBLK 128   // S-rows per workgroup in the scores kernel (8 waves x 16)

typedef __attribute__((ext_vector_type(16))) __bf16        v16bf;
typedef __attribute__((ext_vector_type(16))) unsigned short v16us;
typedef __attribute__((ext_vector_type(8)))  unsigned short v8us;
typedef __attribute__((ext_vector_type(8)))  float          v8f;

#define AS_GLOBAL __attribute__((address_space(1)))
#define AS_LDS    __attribute__((address_space(3)))

#if defined(__has_builtin)
#  if __has_builtin(__builtin_amdgcn_global_load_async_to_lds_b32)
#    define HAVE_ASYNC_BUILTIN 1
#  endif
#  if __has_builtin(__builtin_amdgcn_s_wait_asynccnt)
#    define HAVE_ASYNC_WAIT_BUILTIN 1
#  endif
#endif

// async copy of one dword: global -> LDS (ASYNCcnt-tracked, no VGPR data)
static __device__ __forceinline__ void async_g2l_b32(float* lds, const float* g) {
#if defined(HAVE_ASYNC_BUILTIN)
    __builtin_amdgcn_global_load_async_to_lds_b32(
        (AS_GLOBAL int*)(void*)const_cast<float*>(g),
        (AS_LDS int*)(void*)lds, 0, 0);
#else
    unsigned int loff = (unsigned int)(size_t)(AS_LDS void*)lds;
    asm volatile("global_load_async_to_lds_b32 %0, %1, off"
                 :: "v"(loff), "v"((unsigned long long)(size_t)g)
                 : "memory");
#endif
}

static __device__ __forceinline__ void async_wait0() {
#if defined(HAVE_ASYNC_WAIT_BUILTIN)
    __builtin_amdgcn_s_wait_asynccnt(0);
#else
    asm volatile("s_wait_asynccnt 0x0" ::: "memory");
#endif
}

static __device__ __forceinline__ unsigned short f32_to_bf16_rne(float f) {
    unsigned int u = __float_as_uint(f);
    unsigned int r = 0x7FFFu + ((u >> 16) & 1u);
    return (unsigned short)((u + r) >> 16);
}

// ---------------------------------------------------------------------------
// Kernel A: q_proj[b][n] = dot(query[b][:], Wq[n][:])        grid(32) block(512)
// ---------------------------------------------------------------------------
__global__ void qproj_kernel(const float* __restrict__ query,
                             const float* __restrict__ Wq,
                             float* __restrict__ qproj) {
    __shared__ float qrow[HDIM];
    const int b = blockIdx.x;
    const int n = threadIdx.x;
    qrow[n] = query[b * HDIM + n];
    __syncthreads();
    const float* wr = Wq + (size_t)n * HDIM;
    float acc = 0.f;
#pragma unroll 8
    for (int k = 0; k < HDIM; ++k) acc += qrow[k] * wr[k];
    qproj[b * HDIM + n] = acc;
}

// ---------------------------------------------------------------------------
// Kernel B: fused k_proj GEMM (bf16 WMMA) + tanh + dot(v) + mask -> scores
// grid(S/SBLK, B) block(256).  Dynamic LDS (148 KB -> 2 WGs / 320 KB WGP):
//   kbuf: SBLK x 512 bf16 (128 KB)   keys block
//   wbuf:   16 x 512 bf16 ( 16 KB)   current Wk N-tile
//   qbuf: 512 f32, vbuf: 512 f32     (async-staged)
// ---------------------------------------------------------------------------
__global__ void scores_kernel(const float* __restrict__ keys,
                              const float* __restrict__ Wk,
                              const unsigned char* __restrict__ mask,
                              const float* __restrict__ qproj,
                              const float* __restrict__ v,
                              float* __restrict__ scores) {
    extern __shared__ unsigned char smem[];
    unsigned short* kbuf = (unsigned short*)smem;                         // SBLK*512
    unsigned short* wbuf = (unsigned short*)(smem + SBLK * HDIM * 2);     // 16*512
    float* qbuf = (float*)(smem + SBLK * HDIM * 2 + 16 * HDIM * 2);       // 512
    float* vbuf = qbuf + HDIM;                                            // 512

    const int b    = blockIdx.y;
    const int s0   = blockIdx.x * SBLK;
    const int tid  = threadIdx.x;
    const int lane = tid & 31;
    const int wave = tid >> 5;       // 0..7
    const int lh   = lane & 15;      // column / row-within-tile select
    const int hi   = lane >> 4;      // half-wave: K-chunk select

    // stage q_proj row and v asynchronously (overlaps with keys conversion)
    async_g2l_b32(&qbuf[tid],       qproj + b * HDIM + tid);
    async_g2l_b32(&qbuf[tid + 256], qproj + b * HDIM + tid + 256);
    async_g2l_b32(&vbuf[tid],       v + tid);
    async_g2l_b32(&vbuf[tid + 256], v + tid + 256);

    // stage keys block: SBLK x 512 f32 -> bf16 in LDS
    const float* kb = keys + ((size_t)b * SDIM + s0) * HDIM;
    for (int idx = tid * 4; idx < SBLK * HDIM; idx += 256 * 4) {
        float4 f = *(const float4*)(kb + idx);
        unsigned int p0 = (unsigned)f32_to_bf16_rne(f.x) | ((unsigned)f32_to_bf16_rne(f.y) << 16);
        unsigned int p1 = (unsigned)f32_to_bf16_rne(f.z) | ((unsigned)f32_to_bf16_rne(f.w) << 16);
        *(uint2*)(kbuf + idx) = make_uint2(p0, p1);
    }
    async_wait0();
    __syncthreads();

    float sacc[8];
#pragma unroll
    for (int i = 0; i < 8; ++i) sacc[i] = 0.f;

    // A fragment source row for this lane (rows shared by both half-waves)
    const unsigned short* arow = kbuf + (wave * 16 + lh) * HDIM;

    for (int nt = 0; nt < 32; ++nt) {
        __syncthreads();   // protect wbuf from previous iteration's readers
        // stage Wk tile rows nt*16 .. nt*16+15 (16 x 512 f32 -> bf16)
        const float* wt = Wk + (size_t)nt * 16 * HDIM;
        for (int idx = tid * 4; idx < 16 * HDIM; idx += 256 * 4) {
            float4 f = *(const float4*)(wt + idx);
            unsigned int p0 = (unsigned)f32_to_bf16_rne(f.x) | ((unsigned)f32_to_bf16_rne(f.y) << 16);
            unsigned int p1 = (unsigned)f32_to_bf16_rne(f.z) | ((unsigned)f32_to_bf16_rne(f.w) << 16);
            *(uint2*)(wbuf + idx) = make_uint2(p0, p1);
        }
        // warm caches for the next tile while this one computes (one 128B line/thread)
        if (nt + 1 < 32)
            __builtin_prefetch(Wk + (size_t)(nt + 1) * 16 * HDIM + (size_t)tid * 32, 0, 0);
        __syncthreads();

        const unsigned short* brow = wbuf + lh * HDIM;
        v8f c = {0.f, 0.f, 0.f, 0.f, 0.f, 0.f, 0.f, 0.f};

#pragma unroll
        for (int ks = 0; ks < 16; ++ks) {
            // ISA 16-bit A 16x32 layout: lanes 0-15 K in {0-7,16-23}, lanes 16-31 {8-15,24-31}
            const int k0 = ks * 32 + hi * 8;
            v8us a0 = *(const v8us*)(arow + k0);
            v8us a1 = *(const v8us*)(arow + k0 + 16);
            v8us b0 = *(const v8us*)(brow + k0);
            v8us b1 = *(const v8us*)(brow + k0 + 16);
            v16us au, bu;
#pragma unroll
            for (int i = 0; i < 8; ++i) {
                au[i] = a0[i]; au[i + 8] = a1[i];
                bu[i] = b0[i]; bu[i + 8] = b1[i];
            }
            v16bf av = __builtin_bit_cast(v16bf, au);
            v16bf bv = __builtin_bit_cast(v16bf, bu);
            c = __builtin_amdgcn_wmma_f32_16x16x32_bf16(
                    /*neg_a=*/false, av, /*neg_b=*/false, bv,
                    /*c_mod=*/(short)0, c, /*reuse_a=*/false, /*reuse_b=*/false);
        }

        // fused epilogue: sacc[m-part] += v[n] * tanh(qproj[n] + kproj)
        // C layout: VGPR i -> M = i (lanes 0-15) or i+8 (lanes 16-31); N = lane&15
        const int n  = nt * 16 + lh;
        const float qv = qbuf[n];
        const float vv = vbuf[n];
#pragma unroll
        for (int i = 0; i < 8; ++i) sacc[i] += vv * tanhf(qv + c[i]);
    }

    // reduce over the 16 lanes of each half-wave (N columns) -> per-row score
#pragma unroll
    for (int i = 0; i < 8; ++i) {
        float x = sacc[i];
        x += __shfl_xor(x, 1, 32);
        x += __shfl_xor(x, 2, 32);
        x += __shfl_xor(x, 4, 32);
        x += __shfl_xor(x, 8, 32);
        sacc[i] = x;
    }
    if (lh == 0) {
        const int rbase = s0 + wave * 16 + hi * 8;
#pragma unroll
        for (int i = 0; i < 8; ++i) {
            const int srow = rbase + i;
            const float sc = mask[b * SDIM + srow] ? -1e9f : sacc[i];
            scores[b * SDIM + srow] = sc;
        }
    }
}

// ---------------------------------------------------------------------------
// Kernel C: softmax over S per batch.   grid(32) block(256), 8 elems/thread
// ---------------------------------------------------------------------------
__global__ void softmax_kernel(const float* __restrict__ scores,
                               float* __restrict__ weights) {
    __shared__ float red[256];
    const int b = blockIdx.x, tid = threadIdx.x;
    const float* row = scores + b * SDIM;

    float vals[8];
    float lm = -3.402823466e38f;
#pragma unroll
    for (int i = 0; i < 8; ++i) { vals[i] = row[tid + i * 256]; lm = fmaxf(lm, vals[i]); }
    red[tid] = lm;
    __syncthreads();
    for (int off = 128; off > 0; off >>= 1) {
        if (tid < off) red[tid] = fmaxf(red[tid], red[tid + off]);
        __syncthreads();
    }
    const float m = red[0];
    __syncthreads();

    float ls = 0.f;
#pragma unroll
    for (int i = 0; i < 8; ++i) { vals[i] = __expf(vals[i] - m); ls += vals[i]; }
    red[tid] = ls;
    __syncthreads();
    for (int off = 128; off > 0; off >>= 1) {
        if (tid < off) red[tid] += red[tid + off];
        __syncthreads();
    }
    const float inv = 1.f / red[0];
#pragma unroll
    for (int i = 0; i < 8; ++i) weights[b * SDIM + tid + i * 256] = vals[i] * inv;
}

// ---------------------------------------------------------------------------
// Kernel D: context[b][h] = sum_s weights[b][s] * keys[b][s][h]
// grid(32) block(512); weights async-staged through LDS, keys coalesced over h.
// ---------------------------------------------------------------------------
__global__ void context_kernel(const float* __restrict__ keys,
                               const float* __restrict__ weights,
                               float* __restrict__ context) {
    __shared__ float wch[512];
    const int b = blockIdx.x, tid = threadIdx.x;
    const float* kb = keys + (size_t)b * SDIM * HDIM;
    const float* wr = weights + b * SDIM;
    float acc = 0.f;
    for (int sbase = 0; sbase < SDIM; sbase += 512) {
        __syncthreads();                      // wch free
        async_g2l_b32(&wch[tid], wr + sbase + tid);
        async_wait0();
        __syncthreads();                      // wch ready
#pragma unroll 8
        for (int j = 0; j < 512; ++j)
            acc += wch[j] * kb[(size_t)(sbase + j) * HDIM + tid];
    }
    context[b * HDIM + tid] = acc;
}

// ---------------------------------------------------------------------------
extern "C" void kernel_launch(void* const* d_in, const int* in_sizes, int n_in,
                              void* d_out, int out_size, void* d_ws, size_t ws_size,
                              hipStream_t stream) {
    const float*         query = (const float*)d_in[0];          // (32,512)
    const float*         keys  = (const float*)d_in[1];          // (32,2048,512)
    const unsigned char* mask  = (const unsigned char*)d_in[2];  // (32,2048) bool
    const float*         Wq    = (const float*)d_in[3];          // (512,512)
    const float*         Wk    = (const float*)d_in[4];          // (512,512)
    const float*         v     = (const float*)d_in[5];          // (512,)

    float* out     = (float*)d_out;
    float* context = out;                    // 32*512
    float* weights = out + BDIM * HDIM;      // 32*2048

    float* qproj  = (float*)d_ws;            // 32*512
    float* scores = qproj + BDIM * HDIM;     // 32*2048

    qproj_kernel<<<dim3(BDIM), dim3(HDIM), 0, stream>>>(query, Wq, qproj);

    const size_t lds = (size_t)(SBLK * HDIM * 2 + 16 * HDIM * 2 + HDIM * 4 + HDIM * 4);
    scores_kernel<<<dim3(SDIM / SBLK, BDIM), dim3(256), lds, stream>>>(
        keys, Wk, mask, qproj, v, scores);

    softmax_kernel<<<dim3(BDIM), dim3(256), 0, stream>>>(scores, weights);

    context_kernel<<<dim3(BDIM), dim3(HDIM), 0, stream>>>(keys, weights, context);
}
